// TimeAwareAttention_39032662786127
// MI455X (gfx1250) — matrix operand order repaired and checked by reading, not compiled, not run on hardware
//
#include <hip/hip_runtime.h>
#include <hip/hip_bf16.h>
#include <math.h>

// ---------------------------------------------------------------------------
// TimeAwareAttention for MI455X (gfx1250, wave32, WMMA bf16, async LDS)
// B=8, L=2048, E=1024, D=256, POLY_DEGREE=5
// ---------------------------------------------------------------------------

#define BATCH 8
#define L_SEQ 2048
#define E_DIM 1024
#define D_DIM 256

typedef __attribute__((ext_vector_type(16))) __bf16 v16bf;
typedef __attribute__((ext_vector_type(8)))  float  v8f;

union ABFrag {
    v16bf v;
    uint4 q[2];
};

static __device__ __forceinline__ v8f wmma_bf16(v16bf a, v16bf b, v8f c) {
    // D = A(16x32 bf16) * B(32x16 bf16) + C(16x16 f32)
    return __builtin_amdgcn_wmma_f32_16x16x32_bf16(false, a, false, b, (short)0, c, false, false);
}

// round-to-nearest-even f32 -> bf16 (raw u16)
static __device__ __forceinline__ unsigned short f2bf(float f) {
    unsigned int u = __float_as_uint(f);
    unsigned int r = u + 0x7FFFu + ((u >> 16) & 1u);
    return (unsigned short)(r >> 16);
}

// ---------------------------------------------------------------------------
// Scheduler pipelining helpers (SchedGroupMask: MFMA/WMMA=0x8, VMEM_READ=0x20,
// DS_READ=0x100).  Used to keep fragment loads running ahead of the WMMAs.
// ---------------------------------------------------------------------------
#if __has_builtin(__builtin_amdgcn_sched_group_barrier)
#define SGB(mask, size) __builtin_amdgcn_sched_group_barrier(mask, size, 0)
#else
#define SGB(mask, size)
#endif

// ---------------------------------------------------------------------------
// Async global -> LDS 16B copy (gfx1250 GLOBAL_LOAD_ASYNC_TO_LDS_B128,
// tracked by ASYNCcnt).  Builtin takes (v4i AS1*, v4i AS3*, imm off, imm cpol).
// ---------------------------------------------------------------------------
#if __has_builtin(__builtin_amdgcn_global_load_async_to_lds_b128)
#define HAS_ASYNC_LDS 1
#else
#define HAS_ASYNC_LDS 0
#endif

typedef int v4i_gcc __attribute__((vector_size(16)));
typedef __attribute__((address_space(1))) v4i_gcc* gv4p;
typedef __attribute__((address_space(3))) v4i_gcc* lv4p;

static __device__ __forceinline__ void cp16_g2l(const unsigned short* g, unsigned short* l) {
#if HAS_ASYNC_LDS
    __builtin_amdgcn_global_load_async_to_lds_b128(
        (gv4p)(void*)(unsigned short*)g, (lv4p)(void*)l, 0, 0);
#else
    *(uint4*)l = *(const uint4*)g;
#endif
}

static __device__ __forceinline__ void wait_async0() {
#if HAS_ASYNC_LDS
#if __has_builtin(__builtin_amdgcn_s_wait_asynccnt)
    __builtin_amdgcn_s_wait_asynccnt(0);
#else
    asm volatile("s_wait_asynccnt 0x0" ::: "memory");
#endif
#endif
}

// ---------------------------------------------------------------------------
// Stage 0a: X (f32) -> Xb (bf16), elementwise
// ---------------------------------------------------------------------------
__global__ __launch_bounds__(256) void k_cvt_bf16(const float* __restrict__ in,
                                                  unsigned short* __restrict__ out,
                                                  int n) {
    int i = blockIdx.x * 256 + threadIdx.x;
    if (i < n) out[i] = f2bf(in[i]);
}

// ---------------------------------------------------------------------------
// Stage 0b: W [E,D] f32 -> Wt [D,E] bf16 (transpose + convert)
// ---------------------------------------------------------------------------
__global__ __launch_bounds__(256) void k_wt(const float* __restrict__ W,
                                            unsigned short* __restrict__ Wt) {
    int i = blockIdx.x * 256 + threadIdx.x;   // [0, E*D)
    int k = i >> 8;                            // E index
    int n = i & 255;                           // D index
    Wt[(size_t)n * E_DIM + k] = f2bf(W[i]);
}

// ---------------------------------------------------------------------------
// Stage 1: projection GEMM + exact-erf GELU.
//   One wave computes a 16(M) x 256(N) tile -> X read exactly once.
//   Sched-group pipeline keeps fragment loads ~2 tiles ahead of the WMMAs.
//   vtrans=1 stores output transposed per batch: out[b][d][l] (for V).
// ---------------------------------------------------------------------------
__global__ __launch_bounds__(256) void k_qkv(const unsigned short* __restrict__ Xb,
                                             const unsigned short* __restrict__ Wt,
                                             const float* __restrict__ bias,
                                             unsigned short* __restrict__ outp,
                                             int vtrans) {
    const int lane = threadIdx.x & 31;
    const int mt   = blockIdx.x * 8 + (threadIdx.x >> 5);   // [0, 1024) M tiles
    const int half = lane >> 4;
    const int l16  = lane & 15;

    const unsigned short* xp = Xb + (size_t)(mt * 16 + l16) * E_DIM + half * 8;
    const unsigned short* wp = Wt + (size_t)l16 * E_DIM + half * 8;  // n = l16 base

    v8f acc[16] = {};

    auto loadA = [&](ABFrag& f, int kbv) {
        f.q[0] = *(const uint4*)(xp + kbv);
        f.q[1] = *(const uint4*)(xp + kbv + 16);
    };

    auto phase = [&](ABFrag& aC, ABFrag& aN, int kcur) {
        // Unconditional A prefetch: the final overread lands in workspace.
        loadA(aN, kcur + 32);
        #pragma unroll
        for (int t = 0; t < 16; ++t) {
            const unsigned short* p = wp + (size_t)t * 16 * E_DIM + kcur;
            ABFrag bf;
            bf.q[0] = *(const uint4*)(p);
            bf.q[1] = *(const uint4*)(p + 16);
            acc[t] = wmma_bf16(aC.v, bf.v, acc[t]);
        }
    };

    ABFrag a0, a1;
    loadA(a0, 0);

    for (int kbv = 0; kbv < E_DIM; kbv += 64) {
        phase(a0, a1, kbv);
        phase(a1, a0, kbv + 32);
        // Pipeline the loop body: 68 VMEM reads / 32 WMMAs per iteration.
        // Prime 6 loads, then steady-state 1 WMMA : 2 loads.
        SGB(0x020, 6);
        #pragma unroll
        for (int i = 0; i < 31; ++i) {
            SGB(0x008, 1);
            SGB(0x020, 2);
        }
        SGB(0x008, 1);
    }

    #pragma unroll
    for (int t = 0; t < 16; ++t) {
        int n = t * 16 + l16;
        float bv = bias[n];
        #pragma unroll
        for (int r = 0; r < 8; ++r) {
            float x = acc[t][r] + bv;
            float g = 0.5f * x * (1.0f + erff(x * 0.70710678118654752f));
            int gr = mt * 16 + r + half * 8;            // global row
            unsigned short h = f2bf(g);
            if (vtrans == 0) {
                outp[(size_t)gr * D_DIM + n] = h;
            } else {
                int b = gr >> 11, rl = gr & (L_SEQ - 1);
                outp[((size_t)b * D_DIM + n) * L_SEQ + rl] = h;
            }
        }
    }
}

// ---------------------------------------------------------------------------
// Stage 2: fused temporal flash-attention with block-shared LDS K/V.
//   All 8 waves of a block share one batch -> K/V chunks staged to LDS once
//   per block (async, double-buffered).  One wave: 16 query rows x D=256.
// ---------------------------------------------------------------------------
__global__ __launch_bounds__(256) void k_attn(const unsigned short* __restrict__ qb,
                                              const unsigned short* __restrict__ kmat,
                                              const unsigned short* __restrict__ vtb,
                                              const float* __restrict__ dT,
                                              const unsigned char* __restrict__ maskp,
                                              const float* __restrict__ coeffs,
                                              float* __restrict__ outp) {
    __shared__ __align__(16) unsigned short ldsK[2][32 * 256];  // 2 x 16KB keys
    __shared__ __align__(16) unsigned short ldsV[2][256 * 32];  // 2 x 16KB values (d-major)
    __shared__ __align__(16) unsigned short plds[8][16 * 32];   // 1KB per wave P tiles

    const int tid   = threadIdx.x;
    const int lane  = tid & 31;
    const int wslot = tid >> 5;
    const int wave  = blockIdx.x * 8 + wslot;    // [0, 1024): B*L/16 strips
    const int b     = wave >> 7;                 // 128 strips per batch (uniform per block)
    const int m0    = (wave & 127) << 4;         // query row base in batch
    const int half  = lane >> 4;
    const int l16   = lane & 15;

    const float c0 = coeffs[0], c1 = coeffs[1], c2 = coeffs[2],
                c3 = coeffs[3], c4 = coeffs[4], c5 = coeffs[5];
    const float scale = 0.0625f;                 // 1/sqrt(D)

    const v8f zero = {};

    const unsigned short* kbase = kmat + (size_t)b * L_SEQ * D_DIM;
    const unsigned short* vbase = vtb  + (size_t)b * D_DIM * L_SEQ;
    const float*          dbase = dT   + (size_t)b * L_SEQ * L_SEQ;
    const unsigned char*  mkb   = maskp + b * L_SEQ;

    // Cooperative async staging of one 32-key chunk of K and V into buffer s.
    auto stage = [&](int j0, int s) {
        // K: 32 rows x 256 bf16 (512B/row); thread -> 64B segment
        {
            int row = tid >> 3, seg = tid & 7;
            const unsigned short* g = kbase + (size_t)(j0 + row) * D_DIM + seg * 32;
            unsigned short* l = &ldsK[s][row * 256 + seg * 32];
            cp16_g2l(g,      l);
            cp16_g2l(g + 8,  l + 8);
            cp16_g2l(g + 16, l + 16);
            cp16_g2l(g + 24, l + 24);
        }
        // V (d-major): 256 rows x 32 bf16 (64B/row); thread -> one d row
        {
            const unsigned short* g = vbase + (size_t)tid * L_SEQ + j0;
            unsigned short* l = &ldsV[s][tid * 32];
            cp16_g2l(g,      l);
            cp16_g2l(g + 8,  l + 8);
            cp16_g2l(g + 16, l + 16);
            cp16_g2l(g + 24, l + 24);
        }
    };

    // Cache Q fragments for the whole strip: 8 frags (K=256 in steps of 32)
    ABFrag qf[8];
    {
        const unsigned short* qp =
            qb + (size_t)(b * L_SEQ + m0 + l16) * D_DIM + half * 8;
        #pragma unroll
        for (int kk = 0; kk < 8; ++kk) {
            qf[kk].q[0] = *(const uint4*)(qp + kk * 32);
            qf[kk].q[1] = *(const uint4*)(qp + kk * 32 + 16);
        }
    }

    v8f o[16];
    #pragma unroll
    for (int nt = 0; nt < 16; ++nt) o[nt] = zero;

    float m_i[8], l_i[8];
    #pragma unroll
    for (int r = 0; r < 8; ++r) { m_i[r] = -INFINITY; l_i[r] = 0.0f; }

    stage(0, 0);

    for (int j0 = 0; j0 < L_SEQ; j0 += 32) {
        const int cur = (j0 >> 5) & 1;
        wait_async0();
        __syncthreads();                      // buffer `cur` ready for everyone
        if (j0 + 32 < L_SEQ) stage(j0 + 32, cur ^ 1);

        // delta_t + key mask for the chunk (overlaps S WMMAs)
        float dval[2][8];
        bool  kmv[2];
        #pragma unroll
        for (int jt = 0; jt < 2; ++jt) {
            int kcol = j0 + jt * 16 + l16;
            kmv[jt] = (mkb[kcol] != 0);
            #pragma unroll
            for (int r = 0; r < 8; ++r)
                dval[jt][r] = dbase[(size_t)(m0 + r + half * 8) * L_SEQ + kcol];
        }

        // S = Q * K^T from LDS
        float s[2][8];
        #pragma unroll
        for (int jt = 0; jt < 2; ++jt) {
            const unsigned short* kl =
                &ldsK[cur][(jt * 16 + l16) * 256 + half * 8];
            v8f acc = zero;
            #pragma unroll
            for (int kk = 0; kk < 8; ++kk) {
                ABFrag bf;
                bf.q[0] = *(const uint4*)(kl + kk * 32);
                bf.q[1] = *(const uint4*)(kl + kk * 32 + 16);
                acc = wmma_bf16(qf[kk].v, bf.v, acc);
            }
            #pragma unroll
            for (int r = 0; r < 8; ++r) {
                float d = dval[jt][r];
                float p = c5;
                p = p * d + c4; p = p * d + c3; p = p * d + c2;
                p = p * d + c1; p = p * d + c0;
                float w  = 1.0f / (1.0f + __expf(-p));
                float sc = acc[r] * scale * w;
                s[jt][r] = kmv[jt] ? sc : -INFINITY;
            }
        }

        // Online softmax: row reductions across 16-lane groups (C layout)
        v8f av;
        #pragma unroll
        for (int r = 0; r < 8; ++r) {
            float mx = fmaxf(s[0][r], s[1][r]);
            #pragma unroll
            for (int off = 8; off >= 1; off >>= 1)
                mx = fmaxf(mx, __shfl_xor(mx, off, 16));
            float mnew  = fmaxf(m_i[r], mx);
            float mc    = fmaxf(mnew, -1.0e30f);   // keep exp well-defined if all -inf
            float alpha = __expf(m_i[r] - mc);
            float p0    = __expf(s[0][r] - mc);
            float p1    = __expf(s[1][r] - mc);
            float ls    = p0 + p1;
            #pragma unroll
            for (int off = 8; off >= 1; off >>= 1)
                ls += __shfl_xor(ls, off, 16);
            l_i[r] = l_i[r] * alpha + ls;
            m_i[r] = mnew;
            av[r]  = alpha;
            int prow = r + half * 8;
            plds[wslot][prow * 32 + l16]      = f2bf(p0);
            plds[wslot][prow * 32 + 16 + l16] = f2bf(p1);
        }
        #pragma unroll
        for (int nt = 0; nt < 16; ++nt) o[nt] *= av;

        // In-wave LDS RAW on the P tile: DS ops are in-order; wait + fence
        asm volatile("s_wait_dscnt 0x0" ::: "memory");

        ABFrag pf;
        pf.q[0] = *(const uint4*)(&plds[wslot][l16 * 32 + half * 8]);
        pf.q[1] = *(const uint4*)(&plds[wslot][l16 * 32 + half * 8 + 16]);

        // O += P * V from LDS
        {
            const unsigned short* vl0 = &ldsV[cur][l16 * 32 + half * 8];
            #pragma unroll
            for (int nt = 0; nt < 16; ++nt) {
                const unsigned short* vl = vl0 + (size_t)nt * 16 * 32;
                ABFrag vf;
                vf.q[0] = *(const uint4*)(vl);
                vf.q[1] = *(const uint4*)(vl + 16);
                o[nt] = wmma_bf16(pf.v, vf.v, o[nt]);
            }
            // Pipeline the PV region: 32 DS reads / 16 WMMAs.
            SGB(0x100, 6);
            #pragma unroll
            for (int i = 0; i < 13; ++i) {
                SGB(0x008, 1);
                SGB(0x100, 2);
            }
            SGB(0x008, 3);
        }
    }

    // Epilogue: normalize, honor pad mask, write f32 output [B,L,D]
    #pragma unroll
    for (int r = 0; r < 8; ++r) {
        int  qr = m0 + r + half * 8;
        bool qm = (mkb[qr] != 0);
        float inv = (qm && l_i[r] > 0.0f) ? 1.0f / l_i[r] : 0.0f;
        float* op = outp + ((size_t)b * L_SEQ + qr) * D_DIM + l16;
        #pragma unroll
        for (int nt = 0; nt < 16; ++nt) op[nt * 16] = o[nt][r] * inv;
    }
}

// ---------------------------------------------------------------------------
// Launch
// ---------------------------------------------------------------------------
extern "C" void kernel_launch(void* const* d_in, const int* in_sizes, int n_in,
                              void* d_out, int out_size, void* d_ws, size_t ws_size,
                              hipStream_t stream) {
    const float*         X      = (const float*)d_in[0];
    const float*         dT     = (const float*)d_in[1];
    const unsigned char* mask   = (const unsigned char*)d_in[2];
    const float*         Wq     = (const float*)d_in[3];
    const float*         bq     = (const float*)d_in[4];
    const float*         Wk     = (const float*)d_in[5];
    const float*         bk     = (const float*)d_in[6];
    const float*         Wv     = (const float*)d_in[7];
    const float*         bv     = (const float*)d_in[8];
    const float*         coeffs = (const float*)d_in[9];
    float*               out    = (float*)d_out;

    char* ws = (char*)d_ws;
    size_t off = 0;
    unsigned short* Xb  = (unsigned short*)(ws + off); off += (size_t)BATCH * L_SEQ * E_DIM * 2;
    unsigned short* Wtq = (unsigned short*)(ws + off); off += (size_t)D_DIM * E_DIM * 2;
    unsigned short* Wtk = (unsigned short*)(ws + off); off += (size_t)D_DIM * E_DIM * 2;
    unsigned short* Wtv = (unsigned short*)(ws + off); off += (size_t)D_DIM * E_DIM * 2;
    unsigned short* Qb  = (unsigned short*)(ws + off); off += (size_t)BATCH * L_SEQ * D_DIM * 2;
    unsigned short* Kb  = (unsigned short*)(ws + off); off += (size_t)BATCH * L_SEQ * D_DIM * 2;
    unsigned short* Vtb = (unsigned short*)(ws + off); off += (size_t)BATCH * L_SEQ * D_DIM * 2;
    (void)ws_size; (void)in_sizes; (void)n_in; (void)out_size;

    const int nX = BATCH * L_SEQ * E_DIM;
    k_cvt_bf16<<<(nX + 255) / 256, 256, 0, stream>>>(X, Xb, nX);
    k_wt<<<(E_DIM * D_DIM) / 256, 256, 0, stream>>>(Wq, Wtq);
    k_wt<<<(E_DIM * D_DIM) / 256, 256, 0, stream>>>(Wk, Wtk);
    k_wt<<<(E_DIM * D_DIM) / 256, 256, 0, stream>>>(Wv, Wtv);

    // 1024 waves (one 16x256 tile each), 8 waves per block
    k_qkv<<<128, 256, 0, stream>>>(Xb, Wtq, bq, Qb, 0);
    k_qkv<<<128, 256, 0, stream>>>(Xb, Wtk, bk, Kb, 0);
    k_qkv<<<128, 256, 0, stream>>>(Xb, Wtv, bv, Vtb, 1);

    // 1024 strips (B*L/16), 8 waves per block
    k_attn<<<128, 256, 0, stream>>>(Qb, Kb, Vtb, dT, mask, coeffs, out);
}
